// FFN_19894288515538
// MI455X (gfx1250) — matrix operand order repaired and checked by reading, not compiled, not run on hardware
//
#include <hip/hip_runtime.h>
#include <math.h>

typedef __attribute__((ext_vector_type(16))) _Float16 v16h;
typedef __attribute__((ext_vector_type(8)))  float    v8f;
typedef __attribute__((ext_vector_type(4)))  float    v4f;
typedef __attribute__((ext_vector_type(4)))  unsigned int u32x4;
typedef __attribute__((ext_vector_type(8)))  int      i32x8;
typedef __attribute__((ext_vector_type(4)))  int      i32x4;

#define T_STEPS 4
#define BN_EPS_F 1e-5f

#define BLK_M 128
#define BLK_N 128
#define BLK_K 32

// ---------------------------------------------------------------------------
// Elementwise helpers
// ---------------------------------------------------------------------------

__global__ void f32_to_f16_kernel(const float* __restrict__ in,
                                  _Float16* __restrict__ out, int n) {
  int i = blockIdx.x * blockDim.x + threadIdx.x;
  if (i < n) out[i] = (_Float16)in[i];
}

// IF neuron scan over leading T axis: v += x_t; s = H(v-1); v *= (1-s).
__global__ void if_spike_kernel(const float* __restrict__ x,
                                _Float16* __restrict__ sp, int n) {
  int i = blockIdx.x * blockDim.x + threadIdx.x;
  if (i >= n) return;
  float v = 0.0f;
#pragma unroll
  for (int t = 0; t < T_STEPS; ++t) {
    v += x[(size_t)t * n + i];
    float s = (v >= 1.0f) ? 1.0f : 0.0f;
    sp[(size_t)t * n + i] = (_Float16)s;
    v *= (1.0f - s);
  }
}

// Per-channel (s) mean / rsqrt(var+eps) over (T, L). y: [T, S, L]. One block per s.
__global__ __launch_bounds__(256) void bn_stats_kernel(
    const float* __restrict__ y, float* __restrict__ mean,
    float* __restrict__ rstd, int S, int L) {
  __shared__ float ssum[256];
  __shared__ float ssq[256];
  int s = blockIdx.x;
  float sum = 0.0f, sq = 0.0f;
  for (int t = 0; t < T_STEPS; ++t) {
    const float* p = y + ((size_t)t * S + s) * L;
    for (int l = threadIdx.x; l < L; l += 256) {
      float v = p[l];
      sum += v;
      sq += v * v;
    }
  }
  ssum[threadIdx.x] = sum;
  ssq[threadIdx.x] = sq;
  __syncthreads();
  for (int off = 128; off > 0; off >>= 1) {
    if ((int)threadIdx.x < off) {
      ssum[threadIdx.x] += ssum[threadIdx.x + off];
      ssq[threadIdx.x] += ssq[threadIdx.x + off];
    }
    __syncthreads();
  }
  if (threadIdx.x == 0) {
    float cnt = (float)(T_STEPS * L);
    float m = ssum[0] / cnt;
    float var = ssq[0] / cnt - m * m;  // biased variance
    mean[s] = m;
    rstd[s] = rsqrtf(var + BN_EPS_F);
  }
}

// Fused: normalize then IF scan over T, emit f16 spikes. y:[T,S,L], n = S*L.
__global__ void bn_norm_if_kernel(const float* __restrict__ y,
                                  const float* __restrict__ mean,
                                  const float* __restrict__ rstd,
                                  _Float16* __restrict__ sp, int L, int n) {
  int i = blockIdx.x * blockDim.x + threadIdx.x;
  if (i >= n) return;
  int s = i / L;
  float m = mean[s], r = rstd[s];
  float v = 0.0f;
#pragma unroll
  for (int t = 0; t < T_STEPS; ++t) {
    v += (y[(size_t)t * n + i] - m) * r;
    float sv = (v >= 1.0f) ? 1.0f : 0.0f;
    sp[(size_t)t * n + i] = (_Float16)sv;
    v *= (1.0f - sv);
  }
}

// Final BN normalize in place. y:[T,S,L], n = S*L.
__global__ void bn_norm_kernel(float* __restrict__ y,
                               const float* __restrict__ mean,
                               const float* __restrict__ rstd, int L, int n) {
  int i = blockIdx.x * blockDim.x + threadIdx.x;
  if (i >= n) return;
  int s = i / L;
  float m = mean[s], r = rstd[s];
#pragma unroll
  for (int t = 0; t < T_STEPS; ++t) {
    size_t idx = (size_t)t * n + i;
    y[idx] = (y[idx] - m) * r;
  }
}

// ---------------------------------------------------------------------------
// Tensor Data Mover: DMA a [128 rows x 32 halves] tile (row stride = k_total
// halves) from global memory into LDS at lds_off. 2D D# per CDNA5 ISA §8:
//   group0: count=1 | lds_addr | global_addr(57b) | type=2
//   group1: mask=0, data_size=2B, tensor_dim0=k_total, tensor_dim1=rows_total,
//           tile_dim0=32, tile_dim1=128, tile_dim2=0, dim0_stride=k_total
//   group2/3: zero (tile is 2D, always in-bounds)
// 6-arg builtin form (clang-23 / therock-10.0 headers).
// ---------------------------------------------------------------------------
__device__ __forceinline__ void tdm_load_tile(const _Float16* gptr,
                                              unsigned lds_off, int k_total,
                                              int rows_total) {
  unsigned long long ga = (unsigned long long)(size_t)gptr;
  u32x4 g0;
  g0.x = 1u;                                   // count=1, is_restore=0
  g0.y = lds_off;                              // lds_addr (bytes)
  g0.z = (unsigned)ga;                         // global_addr[31:0]
  g0.w = (unsigned)((ga >> 32) & 0x01ffffffull) | (2u << 30);  // addr[56:32]|type=2
  i32x8 g1;
  g1[0] = 1 << 16;                             // workgroup_mask=0, data_size=2B
  g1[1] = (int)(((unsigned)k_total & 0xffffu) << 16);          // tensor_dim0 lo16
  g1[2] = (int)((((unsigned)k_total >> 16) & 0xffffu) |
                (((unsigned)rows_total & 0xffffu) << 16));     // d0 hi | d1 lo
  g1[3] = (int)((((unsigned)rows_total >> 16) & 0xffffu) |
                ((unsigned)BLK_K << 16));                      // d1 hi | tile_dim0
  g1[4] = BLK_M;                               // tile_dim1=128, tile_dim2=0
  g1[5] = k_total;                             // tensor_dim0_stride lo32
  g1[6] = 0;                                   // stride hi | dim1_stride lo (unused)
  g1[7] = 0;
  i32x4 z4 = {0, 0, 0, 0};
  i32x8 z8 = {0, 0, 0, 0, 0, 0, 0, 0};
  __builtin_amdgcn_tensor_load_to_lds(g0, g1, z4, z4, z8, 0);
}

// ---------------------------------------------------------------------------
// LDS fragment loads matching WMMA VGPR layouts (ISA §7.12.2).
// Tiles stored row-major [128][32] f16.
// ---------------------------------------------------------------------------
__device__ __forceinline__ v16h lds_a_frag(const _Float16* tile, int row,
                                           int hi) {
  const _Float16* p = tile + row * BLK_K + hi * 8;
  union { v16h h; v4f f[2]; } u;
  u.f[0] = *(const v4f*)(p);        // K: hi*8 .. +8
  u.f[1] = *(const v4f*)(p + 16);   // K: 16+hi*8 .. +8
  return u.h;
}

__device__ __forceinline__ v16h lds_b_frag(const _Float16* tile, int col,
                                           int hi) {
  const _Float16* p = tile + col * BLK_K + hi * 8;
  union { v16h h; v4f f[2]; } u;
  u.f[0] = *(const v4f*)(p);        // K: hi*8 .. +8
  u.f[1] = *(const v4f*)(p + 16);   // K: 16+hi*8 .. +8
  return u.h;
}

// ---------------------------------------------------------------------------
// WMMA f16 GEMM with TDM double-buffered LDS staging:
//   C[M,N] = A[M,K] * B[N,K]^T + bias[N]
// Block: 256 threads = 8 waves; block tile 128x128; wave tile 32x64 (2x4
// WMMA 16x16 tiles); K step 32. Wave 0 issues TENSOR_LOAD_TO_LDS for tile
// k+1 while all waves compute tile k from LDS; s_wait_tensorcnt + barrier
// publishes the next buffer.
// ---------------------------------------------------------------------------
__global__ __launch_bounds__(256) void gemm_spike_f16_tdm_kernel(
    const _Float16* __restrict__ A, const _Float16* __restrict__ B,
    const float* __restrict__ bias, float* __restrict__ C,
    int M, int N, int K) {
  __shared__ __align__(16) _Float16 lA[2][BLK_M * BLK_K];
  __shared__ __align__(16) _Float16 lB[2][BLK_N * BLK_K];

  const int lane = threadIdx.x & 31;
  const int wave = threadIdx.x >> 5;
  const int hl = lane & 15;
  const int hi = lane >> 4;
  const int wm = wave >> 1;  // 0..3
  const int wn = wave & 1;   // 0..1
  const int mblk = blockIdx.x * BLK_M;
  const int nblk = blockIdx.y * BLK_N;

  v8f acc[2][4];
#pragma unroll
  for (int i = 0; i < 2; ++i)
#pragma unroll
    for (int j = 0; j < 4; ++j) acc[i][j] = v8f{};

  const int nk = K / BLK_K;

  // Prologue: stage tile 0 into buffer 0.
  if (wave == 0) {
    tdm_load_tile(A + (size_t)mblk * K, (unsigned)(size_t)&lA[0][0], K, M);
    tdm_load_tile(B + (size_t)nblk * K, (unsigned)(size_t)&lB[0][0], K, N);
    __builtin_amdgcn_s_wait_tensorcnt(0);
  }
  __syncthreads();

  int cur = 0;
  for (int ki = 0; ki < nk; ++ki) {
    // Kick off DMA for the next K-tile into the other buffer (overlaps WMMA).
    if (wave == 0 && (ki + 1) < nk) {
      int k = (ki + 1) * BLK_K;
      tdm_load_tile(A + (size_t)mblk * K + k,
                    (unsigned)(size_t)&lA[cur ^ 1][0], K, M);
      tdm_load_tile(B + (size_t)nblk * K + k,
                    (unsigned)(size_t)&lB[cur ^ 1][0], K, N);
    }

    const _Float16* tA = &lA[cur][0];
    const _Float16* tB = &lB[cur][0];
    v16h a0 = lds_a_frag(tA, wm * 32 + hl, hi);
    v16h a1 = lds_a_frag(tA, wm * 32 + 16 + hl, hi);
    v16h b0 = lds_b_frag(tB, wn * 64 + hl, hi);
    v16h b1 = lds_b_frag(tB, wn * 64 + 16 + hl, hi);
    v16h b2 = lds_b_frag(tB, wn * 64 + 32 + hl, hi);
    v16h b3 = lds_b_frag(tB, wn * 64 + 48 + hl, hi);

    acc[0][0] = __builtin_amdgcn_wmma_f32_16x16x32_f16(
        false, a0, false, b0, (short)0, acc[0][0], false, false);
    acc[0][1] = __builtin_amdgcn_wmma_f32_16x16x32_f16(
        false, a0, false, b1, (short)0, acc[0][1], false, false);
    acc[0][2] = __builtin_amdgcn_wmma_f32_16x16x32_f16(
        false, a0, false, b2, (short)0, acc[0][2], false, false);
    acc[0][3] = __builtin_amdgcn_wmma_f32_16x16x32_f16(
        false, a0, false, b3, (short)0, acc[0][3], false, false);
    acc[1][0] = __builtin_amdgcn_wmma_f32_16x16x32_f16(
        false, a1, false, b0, (short)0, acc[1][0], false, false);
    acc[1][1] = __builtin_amdgcn_wmma_f32_16x16x32_f16(
        false, a1, false, b1, (short)0, acc[1][1], false, false);
    acc[1][2] = __builtin_amdgcn_wmma_f32_16x16x32_f16(
        false, a1, false, b2, (short)0, acc[1][2], false, false);
    acc[1][3] = __builtin_amdgcn_wmma_f32_16x16x32_f16(
        false, a1, false, b3, (short)0, acc[1][3], false, false);

    // Ensure the next buffer's DMA has landed before anyone reads it.
    if (wave == 0) __builtin_amdgcn_s_wait_tensorcnt(0);
    __syncthreads();
    cur ^= 1;
  }

  // C/D layout: v8f component j -> (M = tile_m + hi*8 + j, N = tile_n + hl)
  const int m0 = mblk + wm * 32;
  const int n0 = nblk + wn * 64;
#pragma unroll
  for (int mi = 0; mi < 2; ++mi) {
#pragma unroll
    for (int ni = 0; ni < 4; ++ni) {
      int col = n0 + ni * 16 + hl;
      float bv = bias[col];
#pragma unroll
      for (int j = 0; j < 8; ++j) {
        int row = m0 + mi * 16 + hi * 8 + j;
        C[(size_t)row * N + col] = acc[mi][ni][j] + bv;
      }
    }
  }
}

// ---------------------------------------------------------------------------
// Launch
// ---------------------------------------------------------------------------

extern "C" void kernel_launch(void* const* d_in, const int* in_sizes, int n_in,
                              void* d_out, int out_size, void* d_ws,
                              size_t ws_size, hipStream_t stream) {
  (void)in_sizes; (void)n_in; (void)out_size; (void)ws_size;

  const float* x  = (const float*)d_in[0];
  const float* w1 = (const float*)d_in[1];
  const float* b1 = (const float*)d_in[2];
  const float* w2 = (const float*)d_in[3];
  const float* b2 = (const float*)d_in[4];
  float* out = (float*)d_out;

  const int S = 2048, D = 768, H = 3072;
  const int M = T_STEPS * S;  // 8192

  // Workspace carve-out (~173 MB total)
  char* ws = (char*)d_ws;
  size_t off = 0;
  auto carve = [&](size_t bytes) -> void* {
    void* p = ws + off;
    off += (bytes + 255) & ~(size_t)255;
    return p;
  };
  _Float16* sp0   = (_Float16*)carve((size_t)M * D * sizeof(_Float16));
  _Float16* w1h   = (_Float16*)carve((size_t)H * D * sizeof(_Float16));
  _Float16* w2h   = (_Float16*)carve((size_t)D * H * sizeof(_Float16));
  float*    y1    = (float*)carve((size_t)M * H * sizeof(float));
  _Float16* sp1   = (_Float16*)carve((size_t)M * H * sizeof(_Float16));
  float*    mean1 = (float*)carve((size_t)S * sizeof(float));
  float*    rstd1 = (float*)carve((size_t)S * sizeof(float));
  float*    mean2 = (float*)carve((size_t)S * sizeof(float));
  float*    rstd2 = (float*)carve((size_t)S * sizeof(float));

  // 1) weights to f16 (spikes are exactly representable; weights ~5e-4 rel err)
  {
    int n = H * D;
    f32_to_f16_kernel<<<(n + 255) / 256, 256, 0, stream>>>(w1, w1h, n);
    f32_to_f16_kernel<<<(n + 255) / 256, 256, 0, stream>>>(w2, w2h, n);
  }
  // 2) IF spikes of x  -> sp0 [T, S*D]
  {
    int n = S * D;
    if_spike_kernel<<<(n + 255) / 256, 256, 0, stream>>>(x, sp0, n);
  }
  // 3) GEMM1: y1[M,H] = sp0[M,D] * w1h[H,D]^T + b1
  gemm_spike_f16_tdm_kernel<<<dim3(M / BLK_M, H / BLK_N), 256, 0, stream>>>(
      sp0, w1h, b1, y1, M, H, D);
  // 4) BN1 stats per channel s over (T, H)
  bn_stats_kernel<<<S, 256, 0, stream>>>(y1, mean1, rstd1, S, H);
  // 5) normalize + IF -> sp1 [T, S*H]
  {
    int n = S * H;
    bn_norm_if_kernel<<<(n + 255) / 256, 256, 0, stream>>>(y1, mean1, rstd1,
                                                           sp1, H, n);
  }
  // 6) GEMM2: out[M,D] = sp1[M,H] * w2h[D,H]^T + b2
  gemm_spike_f16_tdm_kernel<<<dim3(M / BLK_M, D / BLK_N), 256, 0, stream>>>(
      sp1, w2h, b2, out, M, D, H);
  // 7) BN2 stats per s over (T, D), then normalize d_out in place
  bn_stats_kernel<<<S, 256, 0, stream>>>(out, mean2, rstd2, S, D);
  {
    int n = S * D;
    bn_norm_kernel<<<(n + 255) / 256, 256, 0, stream>>>(out, mean2, rstd2, D,
                                                        n);
  }
}
